// CoarseMatching_77154792506106
// MI455X (gfx1250) — compile-verified, matrix operand order checked by reading, not run on hardware
//
#include <hip/hip_runtime.h>
#include <hip/hip_bf16.h>
#include <math.h>

// ---------------------------------------------------------------------------
// CDNA5 (gfx1250) wave32 WMMA bf16 types
// ---------------------------------------------------------------------------
typedef __attribute__((ext_vector_type(16))) __bf16 v16bf;
typedef __attribute__((ext_vector_type(8)))  __bf16 v8bf;
typedef __attribute__((ext_vector_type(8)))  float  v8f;

#define DEV __device__ __forceinline__

// Operand loader for V_WMMA_F32_16X16X32_BF16 (ISA 7.12.2, 16-bit A 16x32):
//   lane L holds matrix row (A) / column (B) r = base + (L&15),
//   half h = L>>4 selects K-chunks [32s+8h .. +7] and [32s+16+8h .. +7].
//   Row stride Kp is a multiple of 8 -> two 16B-aligned b128 loads.
DEV v16bf load_op(const unsigned short* base_us, int row, int Kp, int lane, int kstep) {
  const __bf16* p = (const __bf16*)base_us +
                    (size_t)(row + (lane & 15)) * (size_t)Kp + 8 * (lane >> 4) + 32 * kstep;
  v8bf lo = *(const v8bf*)p;
  v8bf hi = *(const v8bf*)(p + 16);
  v16bf r;
#pragma unroll
  for (int i = 0; i < 8; ++i) { r[i] = lo[i]; r[i + 8] = hi[i]; }
  return r;
}

DEV v8f wmma_bf16(const v16bf& a, const v16bf& b, const v8f& c) {
  return __builtin_amdgcn_wmma_f32_16x16x32_bf16(false, a, false, b, (short)0, c,
                                                 false, false);
}

// Branchless online logsumexp update of (mx, sm) with one 8-element tile slice:
// one rescale exp + 8 element exps, no exec-mask divergence.
DEV void lse_tile_update(float& mx, float& sm, const v8f& c) {
  float tm = c[0];
#pragma unroll
  for (int j = 1; j < 8; ++j) tm = fmaxf(tm, c[j]);
  float m2 = fmaxf(mx, tm);
  float acc = sm * __expf(mx - m2);
#pragma unroll
  for (int j = 0; j < 8; ++j) acc += __expf(c[j] - m2);
  mx = m2;
  sm = acc;
}

// Phase-1 per-tile processing: online LSE (+ optional fused argmax of
// 2S - collse[s], branchless cndmask/max updates).
template <int MODE>
DEV void p1_process(const v8f& c, int nt, int h,
                    float& mx, float& sm, float& bv, int& bi,
                    const float* __restrict__ collse) {
  lse_tile_update(mx, sm, c);
  if (MODE & 1) {
    const float4* cp = (const float4*)(collse + nt * 16 + 8 * h);
    float4 c0 = cp[0], c1 = cp[1];
    float cl8[8] = {c0.x, c0.y, c0.z, c0.w, c1.x, c1.y, c1.z, c1.w};
#pragma unroll
    for (int j = 0; j < 8; ++j) {
      float t = 2.0f * c[j] - cl8[j];
      int sidx = nt * 16 + 8 * h + j;
      bi = (t > bv) ? sidx : bi;
      bv = fmaxf(bv, t);
    }
  }
}

// Phase-2 per-tile processing: coalesced conf writes.
template <int MODE>
DEV void p2_process(const v8f& c, int nt, int h, int n, int mrow, int Nfull,
                    const float* __restrict__ collse, const float rlj[8],
                    float* __restrict__ conf) {
  float cl = collse[nt * 16 + n];
#pragma unroll
  for (int j = 0; j < 8; ++j) {
    float v = __expf(2.0f * c[j] - cl - rlj[j]);
    if (MODE & 4) { v *= 10.0f; v = (v > 1.0f) ? v : 0.0f; }  // /TEMP + mask
    conf[(size_t)(mrow + j + 8 * h) * (size_t)Nfull + (size_t)(nt * 16 + n)] = v;
  }
}

// ---------------------------------------------------------------------------
// Column pass: one wave owns a 16-column strip of S = A * B^T (B-operand =
// f1 columns cached in VGPRs), ping-pong double-buffered M-tile loop (no
// register copies); lane owns one column, online per-column logsumexp.
// ---------------------------------------------------------------------------
template <int KSTEPS>
__global__ __launch_bounds__(256) void colpass_kernel(
    const unsigned short* __restrict__ A, const unsigned short* __restrict__ B,
    int Mtiles, int Ntiles, int Kp, float* __restrict__ collse) {
  int lane = threadIdx.x & 31;
  int wave = blockIdx.x * 8 + (threadIdx.x >> 5);
  if (wave >= Ntiles) return;
  int ncol = wave * 16;

  v16bf breg[KSTEPS];
#pragma unroll
  for (int s = 0; s < KSTEPS; ++s) breg[s] = load_op(B, ncol, Kp, lane, s);

  float mx = -3.0e38f, sm = 0.0f;

  v16bf buf0[KSTEPS], buf1[KSTEPS];
#pragma unroll
  for (int s = 0; s < KSTEPS; ++s) buf0[s] = load_op(A, 0, Kp, lane, s);

  for (int mt = 0; mt < Mtiles; mt += 2) {
    if (mt + 4 < Mtiles)
      __builtin_prefetch((const __bf16*)A + (size_t)((mt + 4) * 16 + (lane & 15)) * Kp, 0, 1);
    int mtB = (mt + 1 < Mtiles) ? mt + 1 : mt;
#pragma unroll
    for (int s = 0; s < KSTEPS; ++s) buf1[s] = load_op(A, mtB * 16, Kp, lane, s);
    {
      v8f c = {};
#pragma unroll
      for (int s = 0; s < KSTEPS; ++s) c = wmma_bf16(buf0[s], breg[s], c);
      lse_tile_update(mx, sm, c);
    }
    if (mt + 1 < Mtiles) {  // wave-uniform (scalar) branch
      int mtA = (mt + 2 < Mtiles) ? mt + 2 : mt + 1;
#pragma unroll
      for (int s = 0; s < KSTEPS; ++s) buf0[s] = load_op(A, mtA * 16, Kp, lane, s);
      v8f c = {};
#pragma unroll
      for (int s = 0; s < KSTEPS; ++s) c = wmma_bf16(buf1[s], breg[s], c);
      lse_tile_update(mx, sm, c);
    }
  }
  // merge the two half-waves holding rows {j} and {j+8} of the same column
  float mo = __shfl_xor(mx, 16, 32);
  float so = __shfl_xor(sm, 16, 32);
  float m = fmaxf(mx, mo);
  float s2 = sm * __expf(mx - m) + so * __expf(mo - m);
  if (lane < 16) collse[ncol + lane] = m + __logf(s2);
}

// ---------------------------------------------------------------------------
// Row pass: one wave owns a 16-row strip of S.
// Phase 1 computes tiles of S^T (A-op = f1 tile, B-op = cached f0 tile) so
// each lane owns ONE row of S -> branchless per-lane LSE + fused argmax;
// single xor-16 merge. Phase 2 (MODE bit1) swaps operand roles back so conf
// writes are coalesced: conf[l,s] = exp(2S - collse - rowlse).
// MODE bit0: argmax -> idx_out.  MODE bit2: conf12 transform (/TEMP, >1 mask).
// Both phases use the copy-free ping-pong double buffer.
// ---------------------------------------------------------------------------
template <int KSTEPS, int MODE>
__global__ __launch_bounds__(256) void rowpass_kernel(
    const unsigned short* __restrict__ A, const unsigned short* __restrict__ B,
    int Mtiles, int Ntiles, int Kp,
    const float* __restrict__ collse, float* __restrict__ conf,
    int* __restrict__ idx_out, int Nfull) {
  int lane = threadIdx.x & 31;
  int wave = blockIdx.x * 8 + (threadIdx.x >> 5);
  if (wave >= Mtiles) return;
  int mrow = wave * 16;
  int h = lane >> 4, n = lane & 15;

  // cached f0 tile; B-operand in phase 1, A-operand in phase 2 (layouts match)
  v16bf freg[KSTEPS];
#pragma unroll
  for (int s = 0; s < KSTEPS; ++s) freg[s] = load_op(A, mrow, Kp, lane, s);

  float mx = -3.0e38f, sm = 0.0f, bv = -3.0e38f;
  int bi = 0;

  v16bf buf0[KSTEPS], buf1[KSTEPS];
#pragma unroll
  for (int s = 0; s < KSTEPS; ++s) buf0[s] = load_op(B, 0, Kp, lane, s);

  for (int nt = 0; nt < Ntiles; nt += 2) {
    if (nt + 4 < Ntiles)
      __builtin_prefetch((const __bf16*)B + (size_t)((nt + 4) * 16 + n) * Kp, 0, 1);
    int ntB = (nt + 1 < Ntiles) ? nt + 1 : nt;
#pragma unroll
    for (int s = 0; s < KSTEPS; ++s) buf1[s] = load_op(B, ntB * 16, Kp, lane, s);
    {
      v8f c = {};  // c[j] = S[mrow + n, nt*16 + j + 8h]
#pragma unroll
      for (int s = 0; s < KSTEPS; ++s) c = wmma_bf16(buf0[s], freg[s], c);
      p1_process<MODE>(c, nt, h, mx, sm, bv, bi, collse);
    }
    if (nt + 1 < Ntiles) {  // wave-uniform branch
      int ntA = (nt + 2 < Ntiles) ? nt + 2 : nt + 1;
#pragma unroll
      for (int s = 0; s < KSTEPS; ++s) buf0[s] = load_op(B, ntA * 16, Kp, lane, s);
      v8f c = {};
#pragma unroll
      for (int s = 0; s < KSTEPS; ++s) c = wmma_bf16(buf1[s], freg[s], c);
      p1_process<MODE>(c, nt + 1, h, mx, sm, bv, bi, collse);
    }
  }
  // lane n and lane n+16 both own row mrow+n over complementary column sets
  {
    float mo = __shfl_xor(mx, 16, 32);
    float so = __shfl_xor(sm, 16, 32);
    float m = fmaxf(mx, mo);
    sm = sm * __expf(mx - m) + so * __expf(mo - m);
    mx = m;
    if (MODE & 1) {
      float bo = __shfl_xor(bv, 16, 32);
      int io = __shfl_xor(bi, 16, 32);
      bool take = (bo > bv) || (bo == bv && io < bi);
      bv = take ? bo : bv;
      bi = take ? io : bi;
    }
  }
  float rl = mx + __logf(sm);  // rowlse for row mrow + n (valid in all lanes)

  if ((MODE & 1) && lane < 16) idx_out[mrow + lane] = bi;

  if (MODE & 2) {
    // redistribute rowlse to D-layout rows j + 8h
    float rlj[8];
#pragma unroll
    for (int j = 0; j < 8; ++j) rlj[j] = __shfl(rl, j + 8 * h, 32);

#pragma unroll
    for (int s = 0; s < KSTEPS; ++s) buf0[s] = load_op(B, 0, Kp, lane, s);
    for (int nt = 0; nt < Ntiles; nt += 2) {
      int ntB = (nt + 1 < Ntiles) ? nt + 1 : nt;
#pragma unroll
      for (int s = 0; s < KSTEPS; ++s) buf1[s] = load_op(B, ntB * 16, Kp, lane, s);
      {
        v8f c = {};  // c[j] = S[mrow + j + 8h, nt*16 + n]
#pragma unroll
        for (int s = 0; s < KSTEPS; ++s) c = wmma_bf16(freg[s], buf0[s], c);
        p2_process<MODE>(c, nt, h, n, mrow, Nfull, collse, rlj, conf);
      }
      if (nt + 1 < Ntiles) {
        int ntA = (nt + 2 < Ntiles) ? nt + 2 : nt + 1;
#pragma unroll
        for (int s = 0; s < KSTEPS; ++s) buf0[s] = load_op(B, ntA * 16, Kp, lane, s);
        v8f c = {};
#pragma unroll
        for (int s = 0; s < KSTEPS; ++s) c = wmma_bf16(freg[s], buf1[s], c);
        p2_process<MODE>(c, nt + 1, h, n, mrow, Nfull, collse, rlj, conf);
      }
    }
  }
}

// ---------------------------------------------------------------------------
// f32 -> bf16 (RNE) with zero K-padding to Kp (multiple of 32)
// ---------------------------------------------------------------------------
__global__ void cvt_bf16_pad(const float* __restrict__ src, unsigned short* __restrict__ dst,
                             int rows, int K, int Kp) {
  int t = blockIdx.x * blockDim.x + threadIdx.x;
  int total = rows * Kp;
  if (t >= total) return;
  int r = t / Kp, k = t - r * Kp;
  unsigned short out = 0;
  if (k < K) {
    unsigned u = __float_as_uint(src[(size_t)r * K + k]);
    u += 0x7FFFu + ((u >> 16) & 1u);
    out = (unsigned short)(u >> 16);
  }
  dst[t] = out;
}

// ---------------------------------------------------------------------------
// Glue kernels (tiny workloads; gathers folded into conv loops)
// ---------------------------------------------------------------------------
__global__ void t2conv_kernel(const float* __restrict__ W16, const float* __restrict__ t2,
                              const float* __restrict__ f116_0, const float* __restrict__ f116_1,
                              const int* __restrict__ idx116, float* __restrict__ t2o) {
  int t = blockIdx.x * blockDim.x + threadIdx.x;
  if (t >= 256 * 400) return;
  int o = t / 400, p = t - o * 400;
  const float* w = W16 + o * 768;
  const float* g = f116_1 + (size_t)idx116[p] * 256;
  const float* a = t2 + (size_t)p * 256;
  const float* b = f116_0 + (size_t)p * 256;
  float s = 0.0f;
  for (int c = 0; c < 256; ++c)
    s += w[c] * a[c] + w[256 + c] * b[c] + w[512 + c] * g[c];
  t2o[t] = s;
}

__global__ void f18convpool_kernel(const float* __restrict__ W8, const float* __restrict__ f18_0,
                                   const float* __restrict__ f18_1, const int* __restrict__ idx18,
                                   float* __restrict__ outp) {
  int t = blockIdx.x * blockDim.x + threadIdx.x;
  if (t >= 256 * 100) return;
  int o = t / 100, q = t - o * 100;
  int qy = q / 10, qx = q - qy * 10;
  const float* w = W8 + o * 512;
  float s = 0.0f;
  for (int dy = 0; dy < 4; ++dy)
    for (int dx = 0; dx < 4; ++dx) {
      int p = (qy * 4 + dy) * 40 + qx * 4 + dx;
      const float* a = f18_0 + (size_t)p * 256;
      const float* g = f18_1 + (size_t)idx18[p] * 256;
      for (int c = 0; c < 256; ++c) s += w[c] * a[c] + w[256 + c] * g[c];
    }
  outp[t] = s * (1.0f / 16.0f);
}

__global__ void f14pool_kernel(const float* __restrict__ f14_0, const float* __restrict__ f14_1,
                               const int* __restrict__ idx14, float* __restrict__ outp) {
  int t = blockIdx.x * blockDim.x + threadIdx.x;
  if (t >= 392 * 100) return;
  int c = t / 100, q = t - c * 100;
  int qy = q / 10, qx = q - qy * 10;
  float s = 0.0f;
  for (int dy = 0; dy < 8; ++dy)
    for (int dx = 0; dx < 8; ++dx) {
      int p = (qy * 8 + dy) * 80 + qx * 8 + dx;
      s += (c < 196) ? f14_0[(size_t)p * 196 + c]
                     : f14_1[(size_t)idx14[p] * 196 + (c - 196)];
    }
  outp[t] = s * (1.0f / 64.0f);
}

__global__ void t3c_kernel(const float* __restrict__ t3, const float* __restrict__ t2o,
                           const float* __restrict__ f18p, const float* __restrict__ f14p,
                           float* __restrict__ t3c) {
  int t = blockIdx.x * blockDim.x + threadIdx.x;
  if (t >= 1100 * 100) return;
  int c = t / 100, q = t - c * 100;
  float v;
  if (c < 196) {
    v = t3[(size_t)q * 196 + c];
  } else if (c < 452) {
    int o = c - 196;
    int qy = q / 10, qx = q - (q / 10) * 10;
    float s = 0.0f;
    for (int dy = 0; dy < 2; ++dy)
      for (int dx = 0; dx < 2; ++dx)
        s += t2o[o * 400 + (2 * qy + dy) * 20 + 2 * qx + dx];
    v = s * 0.25f;
  } else if (c < 708) {
    v = f18p[(c - 452) * 100 + q];
  } else {
    v = f14p[(c - 708) * 100 + q];
  }
  t3c[t] = v;
}

__global__ void u3_kernel(const float* __restrict__ Wt3, const float* __restrict__ t3c,
                          float* __restrict__ u3) {
  int t = blockIdx.x * blockDim.x + threadIdx.x;
  if (t >= 128 * 100) return;
  int o = t / 100, q = t - o * 100;
  const float* w = Wt3 + o * 1100;
  float s = 0.0f;
  for (int c = 0; c < 1100; ++c) s += w[c] * t3c[c * 100 + q];
  u3[t] = s;
}

__global__ void t3d_kernel(const float* __restrict__ u3, float* __restrict__ t3d) {
  int t = blockIdx.x * blockDim.x + threadIdx.x;
  if (t >= 128 * 25) return;
  int o = t / 25, r = t - o * 25;
  int ry = r / 5, rx = r - ry * 5;
  float s = 0.0f;
  for (int dy = 0; dy < 2; ++dy)
    for (int dx = 0; dx < 2; ++dx)
      s += u3[o * 100 + (2 * ry + dy) * 10 + 2 * rx + dx];
  t3d[t] = s * 0.25f;
}

__global__ void t4d_kernel(const float* __restrict__ Wt4, const float* __restrict__ t3d,
                           const float* __restrict__ t4, float* __restrict__ t4d) {
  int t = blockIdx.x * blockDim.x + threadIdx.x;
  if (t >= 128 * 25) return;
  int o = t / 25, r = t - o * 25;
  const float* w = Wt4 + o * 256;
  const float* tr = t4 + (size_t)r * 128;
  float s = 0.0f;
  for (int c = 0; c < 128; ++c) s += w[c] * t3d[c * 25 + r] + w[128 + c] * tr[c];
  t4d[t] = s;
}

__global__ __launch_bounds__(256) void topic_kernel(
    const float* __restrict__ f12_0, const float* __restrict__ f12_1,
    const float* __restrict__ t4d, float* __restrict__ out_idx) {
  __shared__ float tseq[25 * 128];
  for (int i = threadIdx.x; i < 25 * 128; i += 256) {
    int k = i >> 7, d = i & 127;
    tseq[i] = t4d[d * 25 + k];
  }
  __syncthreads();
  int m = blockIdx.x * 256 + threadIdx.x;
  if (m >= 12800) return;
  const float* row = (m < 6400) ? (f12_0 + (size_t)m * 128)
                                : (f12_1 + (size_t)(m - 6400) * 128);
  float best = -3.0e38f;
  int bi = 0;
  for (int k = 0; k < 25; ++k) {
    const float* tk = tseq + k * 128;
    float s = 0.0f;
    for (int d = 0; d < 128; ++d) s += row[d] * tk[d];
    if (s > best) { best = s; bi = k; }
  }
  out_idx[m] = (float)bi;
}

// ---------------------------------------------------------------------------
// Launch
// ---------------------------------------------------------------------------
extern "C" void kernel_launch(void* const* d_in, const int* in_sizes, int n_in,
                              void* d_out, int out_size, void* d_ws, size_t ws_size,
                              hipStream_t stream) {
  (void)in_sizes; (void)n_in; (void)out_size; (void)ws_size;
  const float* f12_0 = (const float*)d_in[0];
  const float* f12_1 = (const float*)d_in[1];
  const float* f14_0 = (const float*)d_in[2];
  const float* f14_1 = (const float*)d_in[3];
  const float* f18_0 = (const float*)d_in[4];
  const float* f18_1 = (const float*)d_in[5];
  const float* f116_0 = (const float*)d_in[6];
  const float* f116_1 = (const float*)d_in[7];
  /* d_in[8] = t1 (unused by reference outputs) */
  const float* t2  = (const float*)d_in[9];
  const float* t3  = (const float*)d_in[10];
  const float* t4  = (const float*)d_in[11];
  const float* W16 = (const float*)d_in[12];
  const float* W8  = (const float*)d_in[13];
  const float* Wt3 = (const float*)d_in[14];
  const float* Wt4 = (const float*)d_in[15];

  float* conf18 = (float*)d_out;              // 1600*1600
  float* conf12 = conf18 + 2560000;           // 6400*6400
  float* topic  = conf12 + 40960000;          // 12800

  char* ws = (char*)d_ws;
  size_t off = 0;
  auto alloc = [&](size_t bytes) -> void* {
    off = (off + 255) & ~(size_t)255;
    void* p = ws + off;
    off += bytes;
    return p;
  };

  unsigned short* b12_0  = (unsigned short*)alloc((size_t)6400 * 128 * 2);
  unsigned short* b12_1  = (unsigned short*)alloc((size_t)6400 * 128 * 2);
  unsigned short* b14_0  = (unsigned short*)alloc((size_t)6400 * 224 * 2);
  unsigned short* b14_1  = (unsigned short*)alloc((size_t)6400 * 224 * 2);
  unsigned short* b18_0  = (unsigned short*)alloc((size_t)1600 * 256 * 2);
  unsigned short* b18_1  = (unsigned short*)alloc((size_t)1600 * 256 * 2);
  unsigned short* b116_0 = (unsigned short*)alloc((size_t)400 * 256 * 2);
  unsigned short* b116_1 = (unsigned short*)alloc((size_t)400 * 256 * 2);
  float* cl12  = (float*)alloc(6400 * 4);
  float* cl14  = (float*)alloc(6400 * 4);
  float* cl18  = (float*)alloc(1600 * 4);
  float* cl116 = (float*)alloc(400 * 4);
  int* idx116  = (int*)alloc(400 * 4);
  int* idx18   = (int*)alloc(1600 * 4);
  int* idx14   = (int*)alloc(6400 * 4);
  float* t2o   = (float*)alloc((size_t)256 * 400 * 4);
  float* f18p  = (float*)alloc((size_t)256 * 100 * 4);
  float* f14p  = (float*)alloc((size_t)392 * 100 * 4);
  float* t3c   = (float*)alloc((size_t)1100 * 100 * 4);
  float* u3    = (float*)alloc((size_t)128 * 100 * 4);
  float* t3d   = (float*)alloc((size_t)128 * 25 * 4);
  float* t4d   = (float*)alloc((size_t)128 * 25 * 4);

  // 1) f32 -> bf16 (padded K)
  cvt_bf16_pad<<<(6400 * 128 + 255) / 256, 256, 0, stream>>>(f12_0, b12_0, 6400, 128, 128);
  cvt_bf16_pad<<<(6400 * 128 + 255) / 256, 256, 0, stream>>>(f12_1, b12_1, 6400, 128, 128);
  cvt_bf16_pad<<<(6400 * 224 + 255) / 256, 256, 0, stream>>>(f14_0, b14_0, 6400, 196, 224);
  cvt_bf16_pad<<<(6400 * 224 + 255) / 256, 256, 0, stream>>>(f14_1, b14_1, 6400, 196, 224);
  cvt_bf16_pad<<<(1600 * 256 + 255) / 256, 256, 0, stream>>>(f18_0, b18_0, 1600, 256, 256);
  cvt_bf16_pad<<<(1600 * 256 + 255) / 256, 256, 0, stream>>>(f18_1, b18_1, 1600, 256, 256);
  cvt_bf16_pad<<<(400 * 256 + 255) / 256, 256, 0, stream>>>(f116_0, b116_0, 400, 256, 256);
  cvt_bf16_pad<<<(400 * 256 + 255) / 256, 256, 0, stream>>>(f116_1, b116_1, 400, 256, 256);

  // 2) S116: argmax only (MODE=1)
  colpass_kernel<8><<<(25 + 7) / 8, 256, 0, stream>>>(b116_0, b116_1, 25, 25, 256, cl116);
  rowpass_kernel<8, 1><<<(25 + 7) / 8, 256, 0, stream>>>(b116_0, b116_1, 25, 25, 256,
                                                         cl116, nullptr, idx116, 400);
  // 3) S18: conf18 + argmax (MODE=3)
  colpass_kernel<8><<<(100 + 7) / 8, 256, 0, stream>>>(b18_0, b18_1, 100, 100, 256, cl18);
  rowpass_kernel<8, 3><<<(100 + 7) / 8, 256, 0, stream>>>(b18_0, b18_1, 100, 100, 256,
                                                          cl18, conf18, idx18, 1600);
  // 4) S12: conf12 with /TEMP + mask (MODE=6)
  colpass_kernel<4><<<(400 + 7) / 8, 256, 0, stream>>>(b12_0, b12_1, 400, 400, 128, cl12);
  rowpass_kernel<4, 6><<<(400 + 7) / 8, 256, 0, stream>>>(b12_0, b12_1, 400, 400, 128,
                                                          cl12, conf12, nullptr, 6400);
  // 5) S14: argmax only (MODE=1)
  colpass_kernel<7><<<(400 + 7) / 8, 256, 0, stream>>>(b14_0, b14_1, 400, 400, 224, cl14);
  rowpass_kernel<7, 1><<<(400 + 7) / 8, 256, 0, stream>>>(b14_0, b14_1, 400, 400, 224,
                                                          cl14, nullptr, idx14, 6400);

  // 6) glue chain -> t4d
  t2conv_kernel<<<(256 * 400 + 255) / 256, 256, 0, stream>>>(W16, t2, f116_0, f116_1, idx116, t2o);
  f18convpool_kernel<<<(256 * 100 + 255) / 256, 256, 0, stream>>>(W8, f18_0, f18_1, idx18, f18p);
  f14pool_kernel<<<(392 * 100 + 255) / 256, 256, 0, stream>>>(f14_0, f14_1, idx14, f14p);
  t3c_kernel<<<(1100 * 100 + 255) / 256, 256, 0, stream>>>(t3, t2o, f18p, f14p, t3c);
  u3_kernel<<<(128 * 100 + 255) / 256, 256, 0, stream>>>(Wt3, t3c, u3);
  t3d_kernel<<<(128 * 25 + 255) / 256, 256, 0, stream>>>(u3, t3d);
  t4d_kernel<<<(128 * 25 + 255) / 256, 256, 0, stream>>>(Wt4, t3d, t4, t4d);

  // 7) topic argmax over 25 topics
  topic_kernel<<<50, 256, 0, stream>>>(f12_0, f12_1, t4d, topic);
}
